// SpikingConv2DLayer_custom_11381663334464
// MI455X (gfx1250) — matrix operand order repaired and checked by reading, not compiled, not run on hardware
//
#include <hip/hip_runtime.h>
#include <hip/hip_bf16.h>

typedef __attribute__((ext_vector_type(16))) _Float16 v16h;
typedef __attribute__((ext_vector_type(8)))  float    v8f;

#define B_     32
#define CIN    16
#define COUT   32
#define HH     64
#define WW     64
#define NSTEPS 20
#define EPSF   1e-8f

// ---------------------------------------------------------------------------
// prep: d[c][e] = <w_c, w_e> (f16, B-operand for WMMA), inv_norm = 1/(diag+eps),
// zero the per-step spike counters. One 1024-thread block: thread = (c,e).
// ---------------------------------------------------------------------------
__global__ void prep_kernel(const float* __restrict__ w,
                            _Float16* __restrict__ d16,
                            float* __restrict__ inv_norm,
                            unsigned* __restrict__ cnts) {
  const int t = threadIdx.x;        // 0..1023
  const int c = t >> 5, e = t & 31;
  float acc = 0.f;
  for (int i = 0; i < CIN * 9; ++i)
    acc = fmaf(w[c * CIN * 9 + i], w[e * CIN * 9 + i], acc);
  d16[c * 32 + e] = (_Float16)acc;
  if (c == e) inv_norm[e] = 1.0f / (acc + EPSF);
  if (t < NSTEPS) cnts[t] = 0u;
}

// ---------------------------------------------------------------------------
// conv: direct 3x3 SAME conv, one output element per thread (one-time cost).
// ---------------------------------------------------------------------------
__global__ __launch_bounds__(256)
void conv_kernel(const float* __restrict__ x, const float* __restrict__ w,
                 float* __restrict__ convx) {
  const int idx  = blockIdx.x * 256 + threadIdx.x;
  const int wcol = idx & 63;
  const int hrow = (idx >> 6) & 63;
  const int co   = (idx >> 12) & 31;
  const int bb   = idx >> 17;
  const float* wp = w + (size_t)co * (CIN * 9);
  const float* xp = x + (size_t)bb * CIN * HH * WW;
  float acc = 0.f;
  for (int ci = 0; ci < CIN; ++ci) {
    const float* xc = xp + (size_t)ci * HH * WW;
    const float* wc = wp + ci * 9;
#pragma unroll
    for (int kh = 0; kh < 3; ++kh) {
      const int h = hrow + kh - 1;
      if (h < 0 || h >= HH) continue;
#pragma unroll
      for (int kw = 0; kw < 3; ++kw) {
        const int ww2 = wcol + kw - 1;
        if (ww2 < 0 || ww2 >= WW) continue;
        acc = fmaf(xc[h * WW + ww2], wc[kh * 3 + kw], acc);
      }
    }
  }
  convx[idx] = acc;
}

// ---------------------------------------------------------------------------
// temporal recurrence: one wave owns a 32-pixel x 32-channel tile of `mem`
// for all 20 steps. Per step: spk(f16, via wave-private LDS slice) -> 2 A
// operands (16x32), d in 2 B operands (32x16), 4x v_wmma_f32_16x16x32_f16
// give rst; elementwise LIF update in f32 registers; spikes stored to
// spk_rec and counted (u32 atomics -> deterministic).
// ---------------------------------------------------------------------------
__global__ __launch_bounds__(256)
void snn_temporal_kernel(const float* __restrict__ convx,
                         const _Float16* __restrict__ d16,
                         const float* __restrict__ inv_norm,
                         const float* __restrict__ bias,
                         const float* __restrict__ beta_p,
                         float* __restrict__ out,
                         unsigned* __restrict__ cnts) {
  // per-wave 32 rows (pixels) x 40 halfs (32 ch + pad for LDS bank spread)
  __shared__ __align__(16) _Float16 lds[8 * 32 * 40];

  const int tid  = threadIdx.x;
  const int wave = tid >> 5;
  const int lane = tid & 31;
  const int eLo  = lane & 15;   // N index within a 16-wide WMMA tile
  const int half = lane >> 4;   // C/D layout: M = vgpr + 8*half

  const int gw     = blockIdx.x * 8 + wave;  // 0..4095
  const int bb     = gw >> 7;                // batch
  const int pixblk = (gw & 127) * 32;        // 32 consecutive pixels (half row)
  const int hrow   = pixblk >> 6;
  const int w0     = pixblk & 63;            // 0 or 32

  const float beta = beta_p[0];
  const float omb  = 1.0f - beta;

  float invn[2], bv[2];
#pragma unroll
  for (int t = 0; t < 2; ++t) {
    invn[t] = inv_norm[t * 16 + eLo];
    bv[t]   = bias[t * 16 + eLo];
  }

  union HP { v16h v; uint4 q[2]; };

  // B operand: d (symmetric) tiles; K order used here (same as A): lanes<16 hold K=0..15,
  // lanes>=16 hold K=16..31, N = lane%16, two packed K per VGPR.
  HP Bm[2];
  const int halfK = half * 16;
#pragma unroll
  for (int t = 0; t < 2; ++t) {
    const _Float16* dp = d16 + (size_t)(t * 16 + eLo) * 32 + halfK;
    Bm[t].q[0] = *(const uint4*)(dp);
    Bm[t].q[1] = *(const uint4*)(dp + 8);
  }

  // conv term (pre-scaled by 1-beta) and mem, in C/D register layout:
  // element (pix = w0 + p*16 + v + 8*half, e = t*16 + eLo)
  float cv[2][2][8];
  float mem[2][2][8];
#pragma unroll
  for (int p = 0; p < 2; ++p)
#pragma unroll
    for (int t = 0; t < 2; ++t)
#pragma unroll
      for (int v = 0; v < 8; ++v) {
        const int e   = t * 16 + eLo;
        const int pix = w0 + p * 16 + v + 8 * half;
        cv[p][t][v]  = omb * convx[(((size_t)bb * COUT + e) * HH + hrow) * WW + pix];
        mem[p][t][v] = 0.f;
      }

  _Float16* wlds = lds + wave * (32 * 40);
  // spk(t=0 input) = 0
#pragma unroll
  for (int p = 0; p < 2; ++p)
#pragma unroll
    for (int t = 0; t < 2; ++t)
#pragma unroll
      for (int v = 0; v < 8; ++v)
        wlds[(p * 16 + v + 8 * half) * 40 + t * 16 + eLo] = (_Float16)0.f;

  const int kb = half * 16;  // A K order used here (same as B): lanes<16 K=0..15, lanes>=16 K=16..31

  for (int step = 0; step < NSTEPS; ++step) {
    // A operands (spikes) from wave-private LDS slice; M = lane%16
    HP A[2];
#pragma unroll
    for (int p = 0; p < 2; ++p) {
      const _Float16* ap = wlds + (p * 16 + eLo) * 40 + kb;
      A[p].q[0] = *(const uint4*)(ap);
      A[p].q[1] = *(const uint4*)(ap + 8);
    }

    int scount = 0;
#pragma unroll
    for (int p = 0; p < 2; ++p) {
#pragma unroll
      for (int t = 0; t < 2; ++t) {
        v8f rc = {};
        rc = __builtin_amdgcn_wmma_f32_16x16x32_f16(
            false, A[p].v, false, Bm[t].v, (short)0, rc, false, false);
#pragma unroll
        for (int v = 0; v < 8; ++v) {
          const float m = (mem[p][t][v] - rc[v]) * beta + cv[p][t][v];
          mem[p][t][v]  = m;
          const float mthr = m * invn[t] - bv[t];
          const float spk  = (mthr > 0.f) ? 1.0f : 0.0f;
          scount += (mthr > 0.f) ? 1 : 0;
          // feed next step's A
          wlds[(p * 16 + v + 8 * half) * 40 + t * 16 + eLo] = (_Float16)spk;
          // spk_rec layout (B, Cout, T, H, W)
          const int e   = t * 16 + eLo;
          const int pix = w0 + p * 16 + v + 8 * half;
          out[(((size_t)bb * COUT + e) * NSTEPS + step) * (HH * WW)
              + hrow * WW + pix] = spk;
        }
      }
    }
    // deterministic spike counting: wave-reduce, lane0 u32 atomic per step
#pragma unroll
    for (int off = 16; off > 0; off >>= 1)
      scount += __shfl_xor(scount, off, 32);
    if (lane == 0) atomicAdd(&cnts[step], (unsigned)scount);
  }
}

// ---------------------------------------------------------------------------
// finalize: loss = 0.5 * total_spikes / N ; spread = max_t(count_t) / (B*C*H*W)
// ---------------------------------------------------------------------------
__global__ void finalize_kernel(const unsigned* __restrict__ cnts,
                                float* __restrict__ out) {
  if (threadIdx.x == 0) {
    unsigned long long total = 0;
    unsigned mx = 0;
    for (int t = 0; t < NSTEPS; ++t) {
      const unsigned c = cnts[t];
      total += c;
      if (c > mx) mx = c;
    }
    const size_t nspk = (size_t)B_ * COUT * NSTEPS * HH * WW;  // 83,886,080
    out[nspk]     = 0.5f * (float)((double)total / (double)nspk);
    out[nspk + 1] = (float)((double)mx / (double)((size_t)B_ * COUT * HH * WW));
  }
}

extern "C" void kernel_launch(void* const* d_in, const int* in_sizes, int n_in,
                              void* d_out, int out_size, void* d_ws, size_t ws_size,
                              hipStream_t stream) {
  const float* x    = (const float*)d_in[0];  // (32,16,64,64)
  const float* w    = (const float*)d_in[1];  // (32,16,3,3)
  const float* b    = (const float*)d_in[2];  // (32,)
  const float* beta = (const float*)d_in[3];  // (1,)
  // d_in[4] = sigma (unused in forward), d_in[5] = nb_steps (fixed 20)
  float* out = (float*)d_out;

  char*     ws    = (char*)d_ws;
  float*    convx = (float*)ws;                     // 16,777,216 B
  _Float16* d16   = (_Float16*)(ws + 16777216);     //      2,048 B
  float*    invn  = (float*)(ws + 16779264);        //        128 B
  unsigned* cnts  = (unsigned*)(ws + 16779392);     //         80 B

  prep_kernel<<<1, 1024, 0, stream>>>(w, d16, invn, cnts);
  conv_kernel<<<(B_ * COUT * HH * WW) / 256, 256, 0, stream>>>(x, w, convx);
  snn_temporal_kernel<<<512, 256, 0, stream>>>(convx, d16, invn, b, beta, out, cnts);
  finalize_kernel<<<1, 32, 0, stream>>>(cnts, out);
}
